// MoERad_74689481277454
// MI455X (gfx1250) — matrix operand order repaired and checked by reading, not compiled, not run on hardware
//
#include <hip/hip_runtime.h>
#include <hip/hip_bf16.h>

typedef unsigned short u16;
typedef unsigned int   u32;
typedef __attribute__((ext_vector_type(16))) __bf16 v16bf;
typedef __attribute__((ext_vector_type(8)))  float  v8f;

// ---- model constants (from reference) ----
constexpr int cD = 768, cH = 12, cL = 2, cE = 8, cV = 50257;
constexpr int cB = 4, cT = 512, cTI = 49, cIMGF = 2048, cFF = 3072;

// ---- GEMM tiling: 256 threads = 8 waves (wave32); block tile 128x128x32 ----
// 8 waves arranged 4(M) x 2(N); each wave owns 32x64 = 2x4 WMMA 16x16 tiles.
constexpr int BM = 128, BN = 128, BK = 32;

// ------------------------- device helpers -------------------------
__device__ __forceinline__ u16 f2bf(float f) {
  u32 u = __float_as_uint(f);
  u32 r = u + 0x7FFFu + ((u >> 16) & 1u);   // round-to-nearest-even
  return (u16)(r >> 16);
}

__device__ __forceinline__ float gelu_tanh(float x) {
  float x3 = x * x * x;
  return 0.5f * x * (1.f + tanhf(0.7978845608028654f * (x + 0.044715f * x3)));
}

__device__ __forceinline__ float wredsum(float v) {
#pragma unroll
  for (int o = 16; o > 0; o >>= 1) v += __shfl_xor(v, o, 32);
  return v;
}

__device__ __forceinline__ int imin(int a, int b) { return a < b ? a : b; }

// LDS byte-offset of a shared-memory pointer (addrspace(3) is 32-bit,
// value is relative to the wave's LDS base — what async-DMA VDST wants).
__device__ __forceinline__ u32 lds_off(const void* p) {
  return (u32)(size_t)(const __attribute__((address_space(3))) void*)p;
}

// CDNA5 async global->LDS DMA (16B per lane), tracked by ASYNCcnt.
__device__ __forceinline__ void async_copy_b128(u32 lds_byte_off, const void* gptr) {
  asm volatile("global_load_async_to_lds_b128 %0, %1, off"
               :: "v"(lds_byte_off), "v"((unsigned long long)(size_t)gptr)
               : "memory");
}
__device__ __forceinline__ void async_wait_all() {
  asm volatile("s_wait_asynccnt 0x0" ::: "memory");
}

// ------------------------- elementwise kernels -------------------------
// f32 -> bf16, two elements per thread (all our sizes are even)
__global__ void cvt_f32_bf16_k(const float* __restrict__ X, u16* __restrict__ Y, long n2) {
  long i = (long)blockIdx.x * blockDim.x + threadIdx.x;
  if (i >= n2) return;
  float2 v = ((const float2*)X)[i];
  ((u32*)Y)[i] = (u32)f2bf(v.x) | ((u32)f2bf(v.y) << 16);
}

__global__ void embed_k(const int* __restrict__ idx, const float* __restrict__ wte,
                        const float* __restrict__ wpe, float* __restrict__ X, long n) {
  long i = (long)blockIdx.x * blockDim.x + threadIdx.x;
  if (i >= n) return;
  long row = i / cD;
  int d = (int)(i % cD);
  int tok = idx[row];
  X[i] = wte[(size_t)tok * cD + d] + wpe[(size_t)(row % cT) * cD + d];
}

__global__ void addpos_img_k(float* __restrict__ Xi, const float* __restrict__ wpeI, long n) {
  long i = (long)blockIdx.x * blockDim.x + threadIdx.x;
  if (i >= n) return;
  long row = i / cD;
  int d = (int)(i % cD);
  int t = (int)(row % cTI);
  Xi[i] += wpeI[(size_t)t * cD + d];
}

// ------------------------- layernorm: one wave per row -------------------------
__global__ __launch_bounds__(256)
void layernorm_k(const float* __restrict__ X, const float* __restrict__ S,
                 const float* __restrict__ Bb, float* __restrict__ outF,
                 u16* __restrict__ outH, int rows) {
  int lane = threadIdx.x & 31, w = threadIdx.x >> 5;
  int r = blockIdx.x * 8 + w;
  if (r >= rows) return;
  const float* x = X + (size_t)r * cD;
  float s = 0.f, ss = 0.f;
  for (int j = lane; j < cD; j += 32) { float v = x[j]; s += v; ss += v * v; }
  s = wredsum(s); ss = wredsum(ss);
  float mean = s * (1.f / cD);
  float var  = ss * (1.f / cD) - mean * mean;
  float rstd = rsqrtf(var + 1e-5f);
  for (int j = lane; j < cD; j += 32) {
    float y = (x[j] - mean) * rstd * S[j] + Bb[j];
    if (outF) outF[(size_t)r * cD + j] = y;
    outH[(size_t)r * cD + j] = f2bf(y);
  }
}

// ------------------------- MoE gate + top-2: one wave per token -------------------------
__global__ __launch_bounds__(256)
void gate_topk_k(const float* __restrict__ Xf, const float* __restrict__ GW,
                 const float* __restrict__ GB, float* __restrict__ CW, int rows) {
  int lane = threadIdx.x & 31, w = threadIdx.x >> 5;
  int n = blockIdx.x * 8 + w;
  if (n >= rows) return;
  const float* x = Xf + (size_t)n * cD;
  float g[cE];
#pragma unroll
  for (int e = 0; e < cE; ++e) g[e] = 0.f;
  for (int j = lane; j < cD; j += 32) {
    float xv = x[j];
#pragma unroll
    for (int e = 0; e < cE; ++e) g[e] += xv * GW[e * cD + j];
  }
#pragma unroll
  for (int e = 0; e < cE; ++e) {
    g[e] = wredsum(g[e]);
    g[e] = 1.f / (1.f + __expf(-(g[e] + GB[e])));   // sigmoid gate
  }
  if (lane == 0) {
    int i1 = 0; float b1v = g[0];
#pragma unroll
    for (int e = 1; e < cE; ++e) if (g[e] > b1v) { b1v = g[e]; i1 = e; }
    int i2 = -1; float b2v = -1e30f;
#pragma unroll
    for (int e = 0; e < cE; ++e) if (e != i1 && g[e] > b2v) { b2v = g[e]; i2 = e; }
#pragma unroll
    for (int e = 0; e < cE; ++e)
      CW[(size_t)n * cE + e] = (e == i1 || e == i2) ? g[e] : 0.f;
  }
}

// ------------------------- attention: one wave per (b, h, q-row) -------------------------
// hd = 64; each lane owns 2 head dims. Scores in per-wave LDS row
// (same-wave DS ops are in-order on CDNA5 -> no barrier needed).
__global__ __launch_bounds__(256)
void attn_k(const float* __restrict__ Q, const float* __restrict__ Kp,
            const float* __restrict__ Vp, u16* __restrict__ outH,
            int Tq, int Tk, int qStride, int kStride,
            int qOff, int kOff, int vOff, int causal) {
  __shared__ float sc[8][cT];
  int lane = threadIdx.x & 31, w = threadIdx.x >> 5;
  long r = (long)blockIdx.x * 8 + w;
  long total = (long)cB * cH * Tq;
  if (r >= total) return;
  int b = (int)(r / (cH * Tq));
  int rem = (int)(r % (cH * Tq));
  int hh = rem / Tq;
  int t  = rem % Tq;

  const float* qrow = Q + ((size_t)b * Tq + t) * qStride + qOff + hh * 64;
  float q0 = qrow[2 * lane], q1 = qrow[2 * lane + 1];

  float mval = -1e30f;
  for (int k = 0; k < Tk; ++k) {
    const float* krow = Kp + ((size_t)b * Tk + k) * kStride + kOff + hh * 64;
    float p = q0 * krow[2 * lane] + q1 * krow[2 * lane + 1];
    float s = wredsum(p) * 0.125f;            // 1/sqrt(64)
    if (causal && k > t) s = -1e30f;
    if (lane == 0) sc[w][k] = s;
    mval = fmaxf(mval, s);
  }
  float zp = 0.f;
  for (int k = lane; k < Tk; k += 32) {
    float e = __expf(sc[w][k] - mval);
    sc[w][k] = e;
    zp += e;
  }
  float Z = wredsum(zp);
  float a0 = 0.f, a1 = 0.f;
  for (int k = 0; k < Tk; ++k) {
    const float* vrow = Vp + ((size_t)b * Tk + k) * kStride + vOff + hh * 64;
    float a = sc[w][k];
    a0 += a * vrow[2 * lane];
    a1 += a * vrow[2 * lane + 1];
  }
  float inv = 1.f / Z;
  u16* orow = outH + ((size_t)b * Tq + t) * cD + hh * 64;
  orow[2 * lane]     = f2bf(a0 * inv);
  orow[2 * lane + 1] = f2bf(a1 * inv);
}

// ------------------------- WMMA GEMM -------------------------
// out[M,N] = A[M,K](bf16) * W[N,K]^T(bf16), f32 accumulate.
// Double-buffered LDS filled by CDNA5 async global->LDS DMA (ASYNCcnt),
// one workgroup barrier per K-step; 8 v_wmma per wave per K-step.
// Epilogue: +bias[n] | gelu | *rowscale[m] | accumulate-into-outF | bf16 mirror.
__global__ __launch_bounds__(256)
void gemm_bf16_wmma(const u16* __restrict__ A, const u16* __restrict__ W,
                    const float* __restrict__ bias,
                    const float* __restrict__ rs, int rsStride,
                    float* __restrict__ outF, u16* __restrict__ outH,
                    int M, int N, int Kd, int accum, int geluF) {
  __shared__ u16 sA[2][BM * BK];
  __shared__ u16 sB[2][BN * BK];

  const int tid  = threadIdx.x;
  const int lane = tid & 31, wid = tid >> 5;
  const int wmO = (wid >> 1) * 32;     // 4 waves along M
  const int wnO = (wid & 1) * 64;      // 2 waves along N
  const int r16 = lane & 15, hi = lane >> 4;
  const int m0 = blockIdx.y * BM, n0 = blockIdx.x * BN;

  // Per-thread staging: tile = BM*BK bf16 = 8KB = 512 x 16B chunks; 2 per thread.
  // chunk c -> row = c>>2, 16B-group kq = c&3. Row index clamped (not zeroed):
  // clamped rows only feed accumulator rows the epilogue never stores.
  const int c0 = tid, c1 = tid + 256;
  const int rA0 = c0 >> 2, rA1 = c1 >> 2, kq0 = (c0 & 3) * 8, kq1 = (c1 & 3) * 8;
  const u16* gA0 = A + (size_t)imin(m0 + rA0, M - 1) * Kd + kq0;
  const u16* gA1 = A + (size_t)imin(m0 + rA1, M - 1) * Kd + kq1;
  const u16* gB0 = W + (size_t)imin(n0 + rA0, N - 1) * Kd + kq0;
  const u16* gB1 = W + (size_t)imin(n0 + rA1, N - 1) * Kd + kq1;
  u32 lA0[2], lA1[2], lB0[2], lB1[2];
#pragma unroll
  for (int b = 0; b < 2; ++b) {
    lA0[b] = lds_off(&sA[b][rA0 * BK + kq0]);
    lA1[b] = lds_off(&sA[b][rA1 * BK + kq1]);
    lB0[b] = lds_off(&sB[b][rA0 * BK + kq0]);
    lB1[b] = lds_off(&sB[b][rA1 * BK + kq1]);
  }

  v8f acc[2][4] = {};

  // prologue: stage 0 into buffer 0
  async_copy_b128(lA0[0], gA0);
  async_copy_b128(lA1[0], gA1);
  async_copy_b128(lB0[0], gB0);
  async_copy_b128(lB1[0], gB1);

  const int nk = Kd / BK;
  for (int s = 0; s < nk; ++s) {
    async_wait_all();      // my DMA for stage s done
    __syncthreads();       // everyone's DMA visible; prev-stage reads done

    if (s + 1 < nk) {      // prefetch stage s+1 into the other buffer
      const int nb = (s + 1) & 1;
      const size_t ko = (size_t)(s + 1) * BK;
      async_copy_b128(lA0[nb], gA0 + ko);
      async_copy_b128(lA1[nb], gA1 + ko);
      async_copy_b128(lB0[nb], gB0 + ko);
      async_copy_b128(lB1[nb], gB1 + ko);
    }

    // gather fragments per ISA 7.12.2 wave32 layouts and run matrix cores
    const u16* bufA = sA[s & 1];
    const u16* bufB = sB[s & 1];
    union UA { v16bf v; u16 h[16]; } af[2], bfr[4];
#pragma unroll
    for (int j = 0; j < 16; ++j) {
      int ka = 8 * hi + j + ((j >= 8) ? 8 : 0);   // A: 16x32, lane = M%16
      int kb = 8 * hi + j + ((j >= 8) ? 8 : 0);   // B: 32x16, lane = N%16
      af[0].h[j]  = bufA[(wmO + r16) * BK + ka];
      af[1].h[j]  = bufA[(wmO + 16 + r16) * BK + ka];
      bfr[0].h[j] = bufB[(wnO + r16) * BK + kb];
      bfr[1].h[j] = bufB[(wnO + 16 + r16) * BK + kb];
      bfr[2].h[j] = bufB[(wnO + 32 + r16) * BK + kb];
      bfr[3].h[j] = bufB[(wnO + 48 + r16) * BK + kb];
    }
#pragma unroll
    for (int mi = 0; mi < 2; ++mi)
#pragma unroll
      for (int ni = 0; ni < 4; ++ni)
        acc[mi][ni] = __builtin_amdgcn_wmma_f32_16x16x32_bf16(
            false, af[mi].v, false, bfr[ni].v, (short)0, acc[mi][ni], false, false);
  }

  // epilogue: C/D layout -> M = r + 8*hi, N = lane%16
#pragma unroll
  for (int mi = 0; mi < 2; ++mi)
#pragma unroll
    for (int ni = 0; ni < 4; ++ni)
#pragma unroll
      for (int r = 0; r < 8; ++r) {
        int m = m0 + wmO + mi * 16 + hi * 8 + r;
        int n = n0 + wnO + ni * 16 + r16;
        if (m < M && n < N) {
          float v = acc[mi][ni][r];
          if (bias)  v += bias[n];
          if (geluF) v = gelu_tanh(v);
          if (rs)    v *= rs[(size_t)m * rsStride];
          size_t o = (size_t)m * N + n;
          if (outF) outF[o] = accum ? (outF[o] + v) : v;
          if (outH) outH[o] = f2bf(v);
        }
      }
}

// ------------------------- host orchestration -------------------------
static inline size_t alnup(size_t x) { return (x + 255) & ~(size_t)255; }

extern "C" void kernel_launch(void* const* d_in, const int* in_sizes, int n_in,
                              void* d_out, int out_size, void* d_ws, size_t ws_size,
                              hipStream_t stream) {
  (void)in_sizes; (void)n_in; (void)out_size; (void)ws_size;
  const float* img        = (const float*)d_in[0];
  const int*   idx        = (const int*)d_in[1];
  const float* wte        = (const float*)d_in[2];
  const float* wpe        = (const float*)d_in[3];
  const float* wpe_img    = (const float*)d_in[4];
  const float* img_proj_w = (const float*)d_in[5];
  const float* e_ln1_s = (const float*)d_in[6],  *e_ln1_b = (const float*)d_in[7];
  const float* e_qkv_w = (const float*)d_in[8],  *e_qkv_b = (const float*)d_in[9];
  const float* e_proj_w = (const float*)d_in[10], *e_proj_b = (const float*)d_in[11];
  const float* e_ln2_s = (const float*)d_in[12], *e_ln2_b = (const float*)d_in[13];
  const float* e_fc_w  = (const float*)d_in[14], *e_fc_b  = (const float*)d_in[15];
  const float* e_fc2_w = (const float*)d_in[16], *e_fc2_b = (const float*)d_in[17];
  const float* d_ln1_s = (const float*)d_in[18], *d_ln1_b = (const float*)d_in[19];
  const float* d_qkv_w = (const float*)d_in[20], *d_qkv_b = (const float*)d_in[21];
  const float* d_proj_w = (const float*)d_in[22], *d_proj_b = (const float*)d_in[23];
  const float* d_ln2_s = (const float*)d_in[24], *d_ln2_b = (const float*)d_in[25];
  const float* d_kv_w  = (const float*)d_in[26], *d_kv_b  = (const float*)d_in[27];
  const float* d_q_w   = (const float*)d_in[28], *d_q_b   = (const float*)d_in[29];
  const float* d_cproj_w = (const float*)d_in[30], *d_cproj_b = (const float*)d_in[31];
  const float* d_ln3_s = (const float*)d_in[32], *d_ln3_b = (const float*)d_in[33];
  const float* gate_w  = (const float*)d_in[34], *gate_b  = (const float*)d_in[35];
  const float* exp_w1  = (const float*)d_in[36], *exp_b1  = (const float*)d_in[37];
  const float* exp_w2  = (const float*)d_in[38], *exp_b2  = (const float*)d_in[39];
  const float* lnf_s   = (const float*)d_in[40], *lnf_b   = (const float*)d_in[41];
  float* out = (float*)d_out;

  const int Ntxt = cB * cT;   // 2048
  const int Nimg = cB * cTI;  // 196

  // workspace bump allocator (deterministic; identical layout every call)
  char* wsb = (char*)d_ws;
  size_t off = 0;
  auto allocF = [&](size_t n) -> float* { float* p = (float*)(wsb + off); off = alnup(off + n * sizeof(float)); return p; };
  auto allocH = [&](size_t n) -> u16*   { u16*   p = (u16*)(wsb + off);   off = alnup(off + n * sizeof(u16));   return p; };

  // activations
  u16*   imgH  = allocH((size_t)cB * cTI * cIMGF);
  float* xF    = allocF((size_t)Ntxt * cD);
  float* xiF   = allocF((size_t)Nimg * cD);
  float* hF    = allocF((size_t)Ntxt * cD);
  u16*   hH    = allocH((size_t)Ntxt * cD);
  float* qkvF  = allocF((size_t)Ntxt * 3 * cD);
  u16*   attnH = allocH((size_t)Ntxt * cD);
  u16*   ffH   = allocH((size_t)Ntxt * cFF);
  float* kvF   = allocF((size_t)Nimg * 2 * cD);
  float* qF    = allocF((size_t)Ntxt * cD);
  float* cwF   = allocF((size_t)Ntxt * cE);
  // bf16 weight mirrors (converted once per launch; one streaming pass)
  u16* wteH  = allocH((size_t)cV * cD);
  u16* ipjH  = allocH((size_t)cD * cIMGF);
  u16* eqkvH = allocH((size_t)cL * 3 * cD * cD);
  u16* eprjH = allocH((size_t)cL * cD * cD);
  u16* efcH  = allocH((size_t)cL * cFF * cD);
  u16* efc2H = allocH((size_t)cL * cD * cFF);
  u16* dqkvH = allocH((size_t)cL * 3 * cD * cD);
  u16* dprjH = allocH((size_t)cL * cD * cD);
  u16* dkvH  = allocH((size_t)cL * 2 * cD * cD);
  u16* dqH   = allocH((size_t)cL * cD * cD);
  u16* dcpjH = allocH((size_t)cL * cD * cD);
  u16* ew1H  = allocH((size_t)cL * cE * cFF * cD);
  u16* ew2H  = allocH((size_t)cL * cE * cD * cFF);

  auto cvt = [&](const float* src, u16* dst, size_t n) {
    long n2 = (long)(n / 2);
    cvt_f32_bf16_k<<<dim3((unsigned)((n2 + 255) / 256)), dim3(256), 0, stream>>>(src, dst, n2);
  };
  auto gemm = [&](const u16* A, const u16* W, const float* bias, const float* rs, int rsStride,
                  float* oF, u16* oH, int M, int N, int Kd, int accum, int geluF) {
    dim3 g((N + BN - 1) / BN, (M + BM - 1) / BM), blk(256);
    gemm_bf16_wmma<<<g, blk, 0, stream>>>(A, W, bias, rs, rsStride, oF, oH, M, N, Kd, accum, geluF);
  };
  auto ln = [&](const float* X, const float* S, const float* Bb, float* oF, u16* oH, int rows) {
    layernorm_k<<<dim3((rows + 7) / 8), dim3(256), 0, stream>>>(X, S, Bb, oF, oH, rows);
  };
  auto attn = [&](const float* Q, const float* Kp, const float* Vp, u16* o,
                  int Tq, int Tk, int qs, int ks, int qo, int ko, int vo, int causal) {
    long rows = (long)cB * cH * Tq;
    attn_k<<<dim3((unsigned)((rows + 7) / 8)), dim3(256), 0, stream>>>(Q, Kp, Vp, o, Tq, Tk, qs, ks, qo, ko, vo, causal);
  };

  // ---- weight conversion (f32 -> bf16, streamed once) ----
  cvt(wte, wteH, (size_t)cV * cD);
  cvt(img_proj_w, ipjH, (size_t)cD * cIMGF);
  cvt(e_qkv_w, eqkvH, (size_t)cL * 3 * cD * cD);
  cvt(e_proj_w, eprjH, (size_t)cL * cD * cD);
  cvt(e_fc_w, efcH, (size_t)cL * cFF * cD);
  cvt(e_fc2_w, efc2H, (size_t)cL * cD * cFF);
  cvt(d_qkv_w, dqkvH, (size_t)cL * 3 * cD * cD);
  cvt(d_proj_w, dprjH, (size_t)cL * cD * cD);
  cvt(d_kv_w, dkvH, (size_t)cL * 2 * cD * cD);
  cvt(d_q_w, dqH, (size_t)cL * cD * cD);
  cvt(d_cproj_w, dcpjH, (size_t)cL * cD * cD);
  cvt(exp_w1, ew1H, (size_t)cL * cE * cFF * cD);
  cvt(exp_w2, ew2H, (size_t)cL * cE * cD * cFF);
  cvt(img, imgH, (size_t)cB * cTI * cIMGF);

  // ---- embeddings ----
  { long n = (long)Ntxt * cD;
    embed_k<<<dim3((unsigned)((n + 255) / 256)), dim3(256), 0, stream>>>(idx, wte, wpe, xF, n); }
  gemm(imgH, ipjH, nullptr, nullptr, 0, xiF, nullptr, Nimg, cD, cIMGF, 0, 0);
  { long n = (long)Nimg * cD;
    addpos_img_k<<<dim3((unsigned)((n + 255) / 256)), dim3(256), 0, stream>>>(xiF, wpe_img, n); }

  // ---- visual encoder ----
  for (int i = 0; i < cL; ++i) {
    ln(xiF, e_ln1_s + i * cD, e_ln1_b + i * cD, nullptr, hH, Nimg);
    gemm(hH, eqkvH + (size_t)i * 3 * cD * cD, e_qkv_b + (size_t)i * 3 * cD,
         nullptr, 0, qkvF, nullptr, Nimg, 3 * cD, cD, 0, 0);
    attn(qkvF, qkvF, qkvF, attnH, cTI, cTI, 3 * cD, 3 * cD, 0, cD, 2 * cD, 0);
    gemm(attnH, eprjH + (size_t)i * cD * cD, e_proj_b + (size_t)i * cD,
         nullptr, 0, xiF, nullptr, Nimg, cD, cD, 1, 0);
    ln(xiF, e_ln2_s + i * cD, e_ln2_b + i * cD, nullptr, hH, Nimg);
    gemm(hH, efcH + (size_t)i * cFF * cD, e_fc_b + (size_t)i * cFF,
         nullptr, 0, nullptr, ffH, Nimg, cFF, cD, 0, 1);
    gemm(ffH, efc2H + (size_t)i * cD * cFF, e_fc2_b + (size_t)i * cD,
         nullptr, 0, xiF, nullptr, Nimg, cD, cFF, 1, 0);
  }

  // ---- decoder ----
  for (int i = 0; i < cL; ++i) {
    // causal self-attention
    ln(xF, d_ln1_s + i * cD, d_ln1_b + i * cD, nullptr, hH, Ntxt);
    gemm(hH, dqkvH + (size_t)i * 3 * cD * cD, d_qkv_b + (size_t)i * 3 * cD,
         nullptr, 0, qkvF, nullptr, Ntxt, 3 * cD, cD, 0, 0);
    attn(qkvF, qkvF, qkvF, attnH, cT, cT, 3 * cD, 3 * cD, 0, cD, 2 * cD, 1);
    gemm(attnH, dprjH + (size_t)i * cD * cD, d_proj_b + (size_t)i * cD,
         nullptr, 0, xF, nullptr, Ntxt, cD, cD, 1, 0);

    // cross-attention (kv from image tokens, shared ln2 params)
    ln(xiF, d_ln2_s + i * cD, d_ln2_b + i * cD, nullptr, hH, Nimg);
    gemm(hH, dkvH + (size_t)i * 2 * cD * cD, d_kv_b + (size_t)i * 2 * cD,
         nullptr, 0, kvF, nullptr, Nimg, 2 * cD, cD, 0, 0);
    ln(xF, d_ln2_s + i * cD, d_ln2_b + i * cD, nullptr, hH, Ntxt);
    gemm(hH, dqH + (size_t)i * cD * cD, d_q_b + (size_t)i * cD,
         nullptr, 0, qF, nullptr, Ntxt, cD, cD, 0, 0);
    attn(qF, kvF, kvF, attnH, cT, cTI, cD, 2 * cD, 0, 0, cD, 0);
    gemm(attnH, dcpjH + (size_t)i * cD * cD, d_cproj_b + (size_t)i * cD,
         nullptr, 0, xF, nullptr, Ntxt, cD, cD, 1, 0);

    // MoE: top-2 sigmoid gates folded into expert-2 GEMM epilogue
    ln(xF, d_ln3_s + i * cD, d_ln3_b + i * cD, hF, hH, Ntxt);
    gate_topk_k<<<dim3((Ntxt + 7) / 8), dim3(256), 0, stream>>>(
        hF, gate_w + (size_t)i * cE * cD, gate_b + (size_t)i * cE, cwF, Ntxt);
    for (int e = 0; e < cE; ++e) {
      size_t we = (size_t)i * cE + e;
      gemm(hH, ew1H + we * cFF * cD, exp_b1 + we * cFF,
           nullptr, 0, nullptr, ffH, Ntxt, cFF, cD, 0, 1);
      gemm(ffH, ew2H + we * cD * cFF, exp_b2 + we * cD,
           cwF + e, cE, xF, nullptr, Ntxt, cD, cFF, 1, 0);
    }
  }

  // ---- final LN + tied LM head (bf16 weight mirror lives in L2) ----
  ln(xF, lnf_s, lnf_b, nullptr, hH, Ntxt);
  gemm(hH, wteH, nullptr, nullptr, 0, out, nullptr, Ntxt, cV, cD, 0, 0);
}
